// HeteroGraphFeatureExtractor_68839735820634
// MI455X (gfx1250) — matrix operand order repaired and checked by reading, not compiled, not run on hardware
//
#include <hip/hip_runtime.h>
#include <math.h>

#define HDIM 128
#define KTOT 256
#define NEG_SLOPE 0.01f
#define BN_EPS 1e-5f
#define BN_PARTS 240

typedef __attribute__((ext_vector_type(16))) _Float16 v16h;
typedef __attribute__((ext_vector_type(8)))  float    v8f;

// ---------------------------------------------------------------- utilities
__global__ void zero_f32_kernel(float* __restrict__ p, size_t n) {
    size_t i = (size_t)blockIdx.x * blockDim.x + threadIdx.x;
    size_t stride = (size_t)gridDim.x * blockDim.x;
    for (; i < n; i += stride) p[i] = 0.0f;
}

// ------------------------------------------------- edge scatter (mean accum)
// one wave per edge: 32 lanes x float4 == 128 features, fully coalesced.
// atomics stay in L2 (dst arrays are L2-resident: 102MB/26MB << 192MB).
__global__ __launch_bounds__(256)
void scatter_sum_kernel(const float* __restrict__ xsrc, const int* __restrict__ ei,
                        int E, float* __restrict__ agg, float* __restrict__ cnt) {
    long long gid = (long long)blockIdx.x * blockDim.x + threadIdx.x;
    int e    = (int)(gid >> 5);
    int lane = (int)(gid & 31);
    if (e >= E) return;
    int src = ei[e];        // edge_index row 0
    int dst = ei[E + e];    // edge_index row 1
    const float4 v = *(const float4*)(xsrc + (size_t)src * HDIM + lane * 4);
    float* ag = agg + (size_t)dst * HDIM + lane * 4;
    __hip_atomic_fetch_add(ag + 0, v.x, __ATOMIC_RELAXED, __HIP_MEMORY_SCOPE_AGENT);
    __hip_atomic_fetch_add(ag + 1, v.y, __ATOMIC_RELAXED, __HIP_MEMORY_SCOPE_AGENT);
    __hip_atomic_fetch_add(ag + 2, v.z, __ATOMIC_RELAXED, __HIP_MEMORY_SCOPE_AGENT);
    __hip_atomic_fetch_add(ag + 3, v.w, __ATOMIC_RELAXED, __HIP_MEMORY_SCOPE_AGENT);
    if (lane == 0)
        __hip_atomic_fetch_add(cnt + dst, 1.0f, __ATOMIC_RELAXED, __HIP_MEMORY_SCOPE_AGENT);
}

// ------------------------------------- f32 -> f16 combined weight [out][256]
// Wc[o][k] = (k < 128) ? Wl[o][k] : Wr[o][k-128]   (B column o is contiguous)
__global__ void build_wc_kernel(const float* __restrict__ Wl, const float* __restrict__ Wr,
                                _Float16* __restrict__ Wc) {
    int i = blockIdx.x * blockDim.x + threadIdx.x;   // 0..32767
    int o = i >> 8;
    int k = i & 255;
    float v = (k < HDIM) ? Wl[o * HDIM + k] : Wr[o * HDIM + (k - HDIM)];
    Wc[i] = (_Float16)v;
}

// ----------------------------------------------------------- WMMA A fragment
// lane holds K = [g0, g0+8) U [g1, g1+8): v16h element e -> VGPR e/2, half e%2
__device__ __forceinline__ v16h make_afrag(const float* __restrict__ g0,
                                           const float* __restrict__ g1, float s) {
    const float4 u0 = *(const float4*)(g0);
    const float4 u1 = *(const float4*)(g0 + 4);
    const float4 u2 = *(const float4*)(g1);
    const float4 u3 = *(const float4*)(g1 + 4);
    v16h a;
    a[0]  = (_Float16)(u0.x * s); a[1]  = (_Float16)(u0.y * s);
    a[2]  = (_Float16)(u0.z * s); a[3]  = (_Float16)(u0.w * s);
    a[4]  = (_Float16)(u1.x * s); a[5]  = (_Float16)(u1.y * s);
    a[6]  = (_Float16)(u1.z * s); a[7]  = (_Float16)(u1.w * s);
    a[8]  = (_Float16)(u2.x * s); a[9]  = (_Float16)(u2.y * s);
    a[10] = (_Float16)(u2.z * s); a[11] = (_Float16)(u2.w * s);
    a[12] = (_Float16)(u3.x * s); a[13] = (_Float16)(u3.y * s);
    a[14] = (_Float16)(u3.z * s); a[15] = (_Float16)(u3.w * s);
    return a;
}

// -------------------------------------------------------------- fused GEMM
// out[r,:] (+)= 0.5 * ( [agg[r]/max(cnt,1) || xdst[r]] @ Wc^T + bias )
// block = 8 waves; wave w -> 16-col tile w; each wave sweeps 8 x 16-row tiles
// reusing its 8 preloaded B fragments (64 v_wmma_f32_16x16x32_f16 per wave).
__global__ __launch_bounds__(256)
void sage_gemm_wmma_kernel(const float* __restrict__ agg, const float* __restrict__ cnt,
                           const float* __restrict__ xdst, const _Float16* __restrict__ Wc,
                           const float* __restrict__ bias, float* __restrict__ out,
                           int nrows, int accumulate) {
    const int tid  = threadIdx.x;
    const int wave = tid >> 5;
    const int lane = tid & 31;
    const int l16  = lane & 15;
    const int hsel = lane >> 4;
    const int col0 = wave * 16;
    const int block_row0 = blockIdx.x * 128;

    // B fragments: lane (n=l16) holds K in [ks*32 + hsel*16, +16) of column n
    v16h bfrag[8];
    {
        const _Float16* wrow = Wc + (size_t)(col0 + l16) * KTOT;
#pragma unroll
        for (int ks = 0; ks < 8; ++ks)
            bfrag[ks] = *(const v16h*)(wrow + ks * 32 + hsel * 16);
    }
    const float bias_v = bias[col0 + l16];

#pragma unroll
    for (int mt = 0; mt < 8; ++mt) {
        const int row0 = block_row0 + mt * 16;
        int r  = row0 + l16;
        int rc = (r < nrows) ? r : (nrows - 1);       // clamp tail reads
        const float invc = 1.0f / fmaxf(cnt[rc], 1.0f);
        const float* arow = agg  + (size_t)rc * HDIM;
        const float* xrow = xdst + (size_t)rc * HDIM;
        v8f c = {};
#pragma unroll
        for (int ks = 0; ks < 4; ++ks) {              // K 0..127: mean part
            const int kb = ks * 32;
            v16h a = make_afrag(arow + kb + hsel * 16, arow + kb + hsel * 16 + 8, invc);
            c = __builtin_amdgcn_wmma_f32_16x16x32_f16(false, a, false, bfrag[ks],
                                                       (short)0, c, false, false);
        }
#pragma unroll
        for (int ks = 4; ks < 8; ++ks) {              // K 128..255: x_dst part
            const int kb = ks * 32 - HDIM;
            v16h a = make_afrag(xrow + kb + hsel * 16, xrow + kb + hsel * 16 + 8, 1.0f);
            c = __builtin_amdgcn_wmma_f32_16x16x32_f16(false, a, false, bfrag[ks],
                                                       (short)0, c, false, false);
        }
        // C layout: VGPR j, lanes 0-15 -> M=j, lanes 16-31 -> M=j+8; N = l16
        const int col = col0 + l16;
#pragma unroll
        for (int j = 0; j < 8; ++j) {
            const int row = row0 + hsel * 8 + j;
            if (row < nrows) {
                const size_t idx = (size_t)row * HDIM + col;
                const float v = 0.5f * (c[j] + bias_v);
                out[idx] = accumulate ? (out[idx] + v) : v;
            }
        }
    }
}

// ------------------------------------------------------ BatchNorm (2-stage)
__global__ __launch_bounds__(256)
void bn_stats_partial_kernel(const float* __restrict__ v, int nrows,
                             float* __restrict__ psum, float* __restrict__ psq) {
    __shared__ float ssum[256], ssq[256];
    int tid  = threadIdx.x;
    int col  = tid & 127;
    int half = tid >> 7;
    float s = 0.0f, q = 0.0f;
    for (int r = blockIdx.x * 2 + half; r < nrows; r += gridDim.x * 2) {
        float x = v[(size_t)r * HDIM + col];
        s += x; q += x * x;
    }
    ssum[tid] = s; ssq[tid] = q;
    __syncthreads();
    if (tid < 128) {
        psum[blockIdx.x * 128 + tid] = ssum[tid] + ssum[tid + 128];
        psq [blockIdx.x * 128 + tid] = ssq [tid] + ssq [tid + 128];
    }
}

__global__ void bn_stats_final_kernel(const float* __restrict__ psum,
                                      const float* __restrict__ psq,
                                      int nparts, float* __restrict__ stats) {
    int tid = threadIdx.x;                  // 256 threads: [0,128)=sum, [128,256)=sumsq
    int col = tid & 127;
    const float* src = (tid < 128) ? psum : psq;
    float acc = 0.0f;
    for (int b = 0; b < nparts; ++b) acc += src[b * 128 + col];
    stats[tid] = acc;
}

__global__ void bn_apply_kernel(const float* __restrict__ v, const float* __restrict__ stats,
                                const float* __restrict__ gamma, const float* __restrict__ beta,
                                float* __restrict__ out, int nrows) {
    const float inv_n = 1.0f / (float)nrows;
    size_t total  = (size_t)nrows * HDIM;
    size_t stride = (size_t)gridDim.x * blockDim.x;
    for (size_t i = (size_t)blockIdx.x * blockDim.x + threadIdx.x; i < total; i += stride) {
        int col  = (int)(i & (HDIM - 1));
        float mu  = stats[col] * inv_n;
        float var = stats[HDIM + col] * inv_n - mu * mu;
        float x = (v[i] - mu) * rsqrtf(var + BN_EPS) * gamma[col] + beta[col];
        out[i] = (x >= 0.0f) ? x : NEG_SLOPE * x;
    }
}

// ------------------------------------------------------------------- driver
extern "C" void kernel_launch(void* const* d_in, const int* in_sizes, int n_in,
                              void* d_out, int out_size, void* d_ws, size_t ws_size,
                              hipStream_t stream) {
    (void)n_in; (void)out_size; (void)ws_size;
    const int NH = 50000, NF = 200000;
    const float* x_host = (const float*)d_in[0];
    const float* x_flow = (const float*)d_in[1];
    const int* ei_sends = (const int*)d_in[2];
    const int* ei_rev   = (const int*)d_in[3];
    const int* ei_prec  = (const int*)d_in[4];
    const int* ei_reach = (const int*)d_in[5];
    const float* W_l  = (const float*)d_in[6];   // [2,4,128,128]
    const float* b_l  = (const float*)d_in[7];   // [2,4,128]
    const float* W_r  = (const float*)d_in[8];   // [2,4,128,128]
    const float* bn_g = (const float*)d_in[9];   // [2,128]
    const float* bn_b = (const float*)d_in[10];  // [2,128]
    const int E_s  = in_sizes[2] / 2;
    const int E_rv = in_sizes[3] / 2;
    const int E_p  = in_sizes[4] / 2;
    const int E_rc = in_sizes[5] / 2;

    char* w = (char*)d_ws;
    auto alloc = [&](size_t bytes) -> void* {
        void* p = (void*)w;
        w += (bytes + 255) & ~(size_t)255;
        return p;
    };
    float* agg_sends = (float*)alloc((size_t)NF * HDIM * 4);
    float* agg_prec  = (float*)alloc((size_t)NF * HDIM * 4);
    float* agg_rev   = (float*)alloc((size_t)NH * HDIM * 4);
    float* agg_reach = (float*)alloc((size_t)NH * HDIM * 4);
    float* cnt_s  = (float*)alloc((size_t)NF * 4);
    float* cnt_p  = (float*)alloc((size_t)NF * 4);
    float* cnt_rv = (float*)alloc((size_t)NH * 4);
    float* cnt_rc = (float*)alloc((size_t)NH * 4);
    char*  zero_end = w;                               // everything above gets zeroed
    float* tmp_f = (float*)alloc((size_t)NF * HDIM * 4);
    float* tmp_h = (float*)alloc((size_t)NH * HDIM * 4);
    float* cur_f = (float*)alloc((size_t)NF * HDIM * 4);
    float* cur_h = (float*)alloc((size_t)NH * HDIM * 4);
    _Float16* Wc = (_Float16*)alloc((size_t)HDIM * KTOT * 2);
    float* psum  = (float*)alloc((size_t)BN_PARTS * 128 * 4);
    float* psq   = (float*)alloc((size_t)BN_PARTS * 128 * 4);
    float* stats = (float*)alloc(256 * 4);

    float* out_h = (float*)d_out;
    float* out_f = out_h + (size_t)NH * HDIM;

    for (int blk = 0; blk < 2; ++blk) {
        const float* hin = blk ? cur_h : x_host;
        const float* fin = blk ? cur_f : x_flow;
        float* hout = (blk == 1) ? out_h : cur_h;
        float* fout = (blk == 1) ? out_f : cur_f;

        size_t zero_n = (size_t)(zero_end - (char*)agg_sends) / 4;
        zero_f32_kernel<<<2048, 256, 0, stream>>>(agg_sends, zero_n);

        auto scat = [&](const float* xs, const int* ei, int E, float* agg, float* cnt) {
            long long threads = (long long)E * 32;
            int blocks = (int)((threads + 255) / 256);
            scatter_sum_kernel<<<blocks, 256, 0, stream>>>(xs, ei, E, agg, cnt);
        };
        scat(hin, ei_sends, E_s,  agg_sends, cnt_s);
        scat(fin, ei_prec,  E_p,  agg_prec,  cnt_p);
        scat(fin, ei_rev,   E_rv, agg_rev,   cnt_rv);
        scat(fin, ei_reach, E_rc, agg_reach, cnt_rc);

        auto sage = [&](int et, const float* agg, const float* cnt, const float* xd,
                        float* outp, int nrows, int acc) {
            const float* Wl = W_l + (size_t)(blk * 4 + et) * HDIM * HDIM;
            const float* Wr = W_r + (size_t)(blk * 4 + et) * HDIM * HDIM;
            const float* bl = b_l + (size_t)(blk * 4 + et) * HDIM;
            build_wc_kernel<<<(HDIM * KTOT) / 256, 256, 0, stream>>>(Wl, Wr, Wc);
            sage_gemm_wmma_kernel<<<(nrows + 127) / 128, 256, 0, stream>>>(
                agg, cnt, xd, Wc, bl, outp, nrows, acc);
        };
        // edge types: 0 sends(h->f), 1 rev_sends(f->h), 2 precedes(f->f), 3 reaches(f->h)
        sage(0, agg_sends, cnt_s,  fin, tmp_f, NF, 0);
        sage(2, agg_prec,  cnt_p,  fin, tmp_f, NF, 1);
        sage(1, agg_rev,   cnt_rv, hin, tmp_h, NH, 0);
        sage(3, agg_reach, cnt_rc, hin, tmp_h, NH, 1);

        auto bn = [&](const float* v, int nrows, float* outp) {
            bn_stats_partial_kernel<<<BN_PARTS, 256, 0, stream>>>(v, nrows, psum, psq);
            bn_stats_final_kernel<<<1, 256, 0, stream>>>(psum, psq, BN_PARTS, stats);
            bn_apply_kernel<<<2048, 256, 0, stream>>>(v, stats, bn_g + blk * HDIM,
                                                      bn_b + blk * HDIM, outp, nrows);
        };
        bn(tmp_h, NH, hout);
        bn(tmp_f, NF, fout);
    }
}
